// CubEcc2d_66340064854423
// MI455X (gfx1250) — matrix operand order repaired and checked by reading, not compiled, not run on hardware
//
#include <hip/hip_runtime.h>
#include <hip/hip_bf16.h>

// Problem constants (match reference)
#define B_      32
#define C_      16
#define H_      128
#define W_      128
#define RES_    64
#define NCH     (B_ * C_)          // 512 independent channels
#define NPIX    (H_ * W_)          // 16384 pixels per channel
#define THREADS 256                // 8 waves (wave32) per workgroup
#define HIST_PITCH (THREADS + 1)   // 257: bank = (bin + tid) mod 64 -> conflict-free
#define IMG_BYTES  (NPIX * 4)      // 64 KB image staged in LDS
#define HIST_INTS  (RES_ * HIST_PITCH)
#define SMEM_BYTES (IMG_BYTES + HIST_INTS * 4)   // 131328 B < 160 KB -> 2 WG/WGP

typedef unsigned int u32x4 __attribute__((ext_vector_type(4)));
typedef int          i32x4 __attribute__((ext_vector_type(4)));
typedef int          i32x8 __attribute__((ext_vector_type(8)));
typedef _Float16     v16h  __attribute__((ext_vector_type(16)));
typedef float        v8f   __attribute__((ext_vector_type(8)));

__device__ __forceinline__ int ecc_bin(float F) {
    int t = (int)ceilf(F * 63.0f);           // SCALE = (RES-1)/(T_MAX-T_MIN) = 63
    t = (t < 0) ? 0 : t;
    return (t > RES_ - 1) ? (RES_ - 1) : t;
}

// ---------------------------------------------------------------------------
// Phase 1: one workgroup per (b,c) channel.
//   - 8 waves each TDM-load a 16x128 slice of the image into LDS (async,
//     tracked by TENSORcnt) while threads zero the privatized histograms.
//   - Each thread owns a private 64-bin i32 histogram slice in LDS
//     (addr = bin*257 + tid  ->  conflict-free banks, no atomics).
//   - Cell decomposition: vertex (+1), h-edge (-1), v-edge (-1), face (+1),
//     cell value = max over its vertices (lower-star filtration).
//   - Reduce 256 private copies -> 64 bins, store as f16 row in d_ws.
// ---------------------------------------------------------------------------
__global__ __launch_bounds__(THREADS)
void ecc_hist_kernel(const float* __restrict__ x, _Float16* __restrict__ hist)
{
    extern __shared__ char smem[];
    float* img = (float*)smem;
    int*   h   = (int*)(smem + IMG_BYTES);

    const int tid = threadIdx.x;
    const int bc  = blockIdx.x;
    const int wid = tid >> 5;                  // wave id 0..7

    // --- Tensor Data Mover: this wave's 16-row slice, global -> LDS ---
    {
        unsigned long long ga =
            (unsigned long long)(x + (size_t)bc * NPIX + (size_t)wid * 16 * W_);
        // D# group 0 (128b): count=1, lds_addr, global_addr[56:0], type=2
        u32x4 g0;
        g0.x = 1u;                                        // count=1, user mode
        g0.y = (unsigned int)(wid * 16 * W_ * 4);         // lds_addr (bytes)
        g0.z = (unsigned int)(ga & 0xFFFFFFFFull);        // global_addr[31:0]
        g0.w = ((unsigned int)(ga >> 32) & 0x01FFFFFFu)   // global_addr[56:32]
             | (2u << 30);                                // type = 2 ("image")
        // D# group 1 (256b): data_size=4B, tensor 128x16, tile 128x16, stride0=128
        i32x8 g1;
        g1[0] = (2 << 16);                                // data_size = 2 (4 bytes)
        g1[1] = (W_ & 0xFFFF) << 16;                      // tensor_dim0[15:0]
        g1[2] = ((W_ >> 16) & 0xFFFF) | ((16 & 0xFFFF) << 16); // dim0 hi | dim1 lo
        g1[3] = 0 | ((W_ & 0xFFFF) << 16);                // dim1 hi | tile_dim0=128
        g1[4] = 16;                                       // tile_dim1=16, tile_dim2=0
        g1[5] = W_;                                       // tensor_dim0_stride[31:0]
        g1[6] = 0;
        g1[7] = 0;
        i32x4 z4 = {0, 0, 0, 0};                          // 2-D tensor: groups 2/3 unused
        i32x8 z8 = {0, 0, 0, 0, 0, 0, 0, 0};
        __builtin_amdgcn_tensor_load_to_lds(g0, g1, z4, z4, z8, 0);
    }

    // Zero privatized histograms while the DMA is in flight.
    for (int k = tid; k < HIST_INTS; k += THREADS) h[k] = 0;

    __builtin_amdgcn_s_wait_tensorcnt(0);   // this wave's slice landed in LDS
    __syncthreads();                        // all slices + zeroed hist visible

    // --- Accumulate signed cell histogram (lower-star filtration) ---
    for (int idx = tid; idx < NPIX; idx += THREADS) {
        const int i = idx >> 7;
        const int j = idx & (W_ - 1);
        const bool jn = (j < W_ - 1);
        const bool in_ = (i < H_ - 1);

        const float v = img[idx];
        if (v <= 1.0f) h[ecc_bin(v) * HIST_PITCH + tid] += 1;          // vertex +1

        if (jn) {
            const float e = fmaxf(v, img[idx + 1]);
            if (e <= 1.0f) h[ecc_bin(e) * HIST_PITCH + tid] -= 1;      // h-edge -1
        }
        if (in_) {
            const float e = fmaxf(v, img[idx + W_]);
            if (e <= 1.0f) h[ecc_bin(e) * HIST_PITCH + tid] -= 1;      // v-edge -1
        }
        if (jn && in_) {
            const float f = fmaxf(fmaxf(v, img[idx + 1]),
                                  fmaxf(img[idx + W_], img[idx + W_ + 1]));
            if (f <= 1.0f) h[ecc_bin(f) * HIST_PITCH + tid] += 1;      // face +1
        }
    }
    __syncthreads();

    // --- Reduce 256 private copies per bin; emit f16 histogram row ---
    if (tid < RES_) {
        int s = 0;
        #pragma unroll 8
        for (int t = 0; t < THREADS; ++t) s += h[tid * HIST_PITCH + t];
        hist[bc * RES_ + tid] = (_Float16)(float)s;   // |s| << 2048 -> exact in f16
    }
}

// ---------------------------------------------------------------------------
// Phase 2: cumsum over RES as a triangular matmul on the WMMA pipe.
//   out[512,64] = H[512,64] @ T[64,64],  T[k,n] = (k <= n)
//   One wave (32 threads) per 16-row slab; K=64 -> two chained
//   v_wmma_f32_16x16x32_f16 per 16x16 output tile; B generated in-register.
//   VGPR layouts per CDNA5 ISA 7.12.2 (16-bit A 16x32, B 32x16, f32 C/D).
// ---------------------------------------------------------------------------
__global__ __launch_bounds__(32)
void ecc_cumsum_wmma_kernel(const _Float16* __restrict__ hist,
                            float* __restrict__ out)
{
    const int lane  = threadIdx.x;
    const int mtile = blockIdx.x;        // 0..31 (512 rows / 16)
    const int half  = lane >> 4;         // lane group (K-halves for A/B)
    const int lm    = lane & 15;         // M-row for A, N-col for B/C/D
    const int row   = mtile * 16 + lm;

    // A tile: rows of the f16 histogram, elements placed per ISA A-layout:
    //   K(e) = (e&7) + (e>=8 ? 16 : 0) + 8*half
    v16h a0, a1;
    #pragma unroll
    for (int e = 0; e < 16; ++e) {
        const int K = (e & 7) + ((e >> 3) << 4) + 8 * half;
        a0[e] = hist[row * RES_ + K];          // K chunk 0..31
        a1[e] = hist[row * RES_ + 32 + K];     // K chunk 32..63
    }

    #pragma unroll
    for (int ntile = 0; ntile < 4; ++ntile) {
        const int ncol = ntile * 16 + lm;
        // B tile: T[k,n] = (k <= n); k(e) = (e&7) + (e>=8 ? 16 : 0) + 8*half (matches A)
        v16h b0, b1;
        #pragma unroll
        for (int e = 0; e < 16; ++e) {
            const int k0 = (e & 7) + ((e >> 3) << 4) + 8 * half;
            const int k1 = 32 + k0;
            b0[e] = (k0 <= ncol) ? (_Float16)1.0f : (_Float16)0.0f;
            b1[e] = (k1 <= ncol) ? (_Float16)1.0f : (_Float16)0.0f;
        }
        v8f c = {0.f, 0.f, 0.f, 0.f, 0.f, 0.f, 0.f, 0.f};
        c = __builtin_amdgcn_wmma_f32_16x16x32_f16(false, a0, false, b0,
                                                   (short)0, c, false, false);
        c = __builtin_amdgcn_wmma_f32_16x16x32_f16(false, a1, false, b1,
                                                   (short)0, c, false, false);
        // D layout: VGPR r holds M = r + 8*half, N = lm
        #pragma unroll
        for (int r = 0; r < 8; ++r)
            out[(mtile * 16 + r + 8 * half) * RES_ + ntile * 16 + lm] = c[r];
    }
}

extern "C" void kernel_launch(void* const* d_in, const int* in_sizes, int n_in,
                              void* d_out, int out_size, void* d_ws, size_t ws_size,
                              hipStream_t stream)
{
    const float* x    = (const float*)d_in[0];        // [32,16,128,128] f32
    _Float16*    hist = (_Float16*)d_ws;              // [512,64] f16 scratch (64 KB)
    float*       out  = (float*)d_out;                // [32,16,64] f32

    ecc_hist_kernel<<<NCH, THREADS, SMEM_BYTES, stream>>>(x, hist);
    ecc_cumsum_wmma_kernel<<<NCH / 16, 32, 0, stream>>>(hist, out);
}
